// PointTransformer_83391085019941
// MI455X (gfx1250) — compile-verified, matrix operand order checked by reading, not run on hardware
//
#include <hip/hip_runtime.h>
#include <hip/hip_bf16.h>
#include <math.h>

#define DEV __device__ __forceinline__

typedef __bf16 bf16_t;
typedef bf16_t v16bf __attribute__((ext_vector_type(16)));
typedef bf16_t v8bf  __attribute__((ext_vector_type(8)));
typedef float  v8f   __attribute__((ext_vector_type(8)));

// ---------------- problem constants ----------------
#define BB    32
#define GG    512
#define CC    768
#define M1    16416      // 32*513
#define MF    16384      // 32*512
#define QKN   2304
#define MLPH  3072
#define ADPN  64
#define S3D_  2048
#define D2D_  7296       // 32*19*12
#define NV_   6
#define NCL_  128
#define MID_  12
#define NH1_  6
#define NPQ   48         // padded 3*MID
#define KPP   32         // padded MID for proj GEMM

// ---------------- small device helpers ----------------
DEV unsigned short f2bf(float f) {
  unsigned u = __float_as_uint(f);
  unsigned r = u + 0x7FFFu + ((u >> 16) & 1u);
  return (unsigned short)(r >> 16);
}
DEV float geluf(float x) { return 0.5f * x * (1.f + erff(x * 0.70710678118f)); }
DEV float sigmf(float x) { return 1.f / (1.f + __expf(-x)); }
DEV unsigned fenc(float f) {
  unsigned u = __float_as_uint(f);
  return (u & 0x80000000u) ? ~u : (u | 0x80000000u);
}
DEV float fdec(unsigned e) {
  unsigned u = (e & 0x80000000u) ? (e & 0x7FFFFFFFu) : ~e;
  return __uint_as_float(u);
}

// ---------------- weight convert + transpose + pad : Wt[n*KP+k] = W[k*N+n] ----------------
__global__ void k_wt(const float* __restrict__ W, unsigned short* __restrict__ Wt,
                     int K, int N, int KP, int NP) {
  size_t t = (size_t)blockIdx.x * blockDim.x + threadIdx.x;
  if (t >= (size_t)NP * KP) return;
  int n = (int)(t / KP), k = (int)(t % KP);
  float v = (n < N && k < K) ? W[(size_t)k * N + n] : 0.f;
  Wt[t] = f2bf(v);
}

__global__ void k_zero_u16(unsigned short* p, size_t n) {
  size_t t = (size_t)blockIdx.x * blockDim.x + threadIdx.x;
  if (t < n) p[t] = 0;
}

// ---------------- LayerNorm -> bf16 ----------------
__global__ __launch_bounds__(256)
void k_layernorm_bf16(const float* __restrict__ X, const float* __restrict__ g,
                      const float* __restrict__ bvec, unsigned short* __restrict__ outB,
                      int Cn) {
  __shared__ float s1[256], s2[256];
  const int row = blockIdx.x;
  const float* xr = X + (size_t)row * Cn;
  float s = 0.f, ss = 0.f;
  for (int c = threadIdx.x; c < Cn; c += 256) { float v = xr[c]; s += v; ss += v * v; }
  s1[threadIdx.x] = s; s2[threadIdx.x] = ss; __syncthreads();
  for (int o = 128; o > 0; o >>= 1) {
    if (threadIdx.x < o) { s1[threadIdx.x] += s1[threadIdx.x + o]; s2[threadIdx.x] += s2[threadIdx.x + o]; }
    __syncthreads();
  }
  float mean = s1[0] / Cn;
  float var  = s2[0] / Cn - mean * mean;
  float inv  = rsqrtf(var + 1e-5f);
  for (int c = threadIdx.x; c < Cn; c += 256) {
    float v = (xr[c] - mean) * inv * g[c] + bvec[c];
    outB[(size_t)row * Cn + c] = f2bf(v);
  }
}

// ---------------- generic bf16 WMMA GEMM ----------------
// out = act(A[M,K](bf16) @ Wt[N,K]^T(bf16) + bias) + resid ; fp32 accum.
// act: 0 none, 1 gelu(exact), 2 x*sigmoid(1.702x)
__global__ __launch_bounds__(128)
void k_gemm_wmma(const unsigned short* __restrict__ A,
                 const unsigned short* __restrict__ Bt,
                 const float* __restrict__ bias,
                 const float* __restrict__ resid,
                 float* __restrict__ outF,
                 unsigned short* __restrict__ outB,
                 int M, int N, int K, int Ntrue, int act) {
  const int lane = threadIdx.x & 31;
  const int wave = threadIdx.x >> 5;
  const int wm = wave >> 1, wn = wave & 1;
  const int m0 = blockIdx.y * 64 + wm * 32;
  const int n0 = blockIdx.x * 64 + wn * 32;
  const int sel = (lane >> 4) << 3;  // 0 or 8 (K-chunk select per 16-bit A layout)
  const int rA0 = m0 + (lane & 15);
  const int rB0 = n0 + (lane & 15);

  v8f acc[2][2] = {};
  for (int k0 = 0; k0 < K; k0 += 32) {
    v16bf a[2], b[2];
#pragma unroll
    for (int t = 0; t < 2; t++) {
      int ra = rA0 + t * 16; if (ra > M - 1) ra = M - 1;
      const unsigned short* pa = A + (size_t)ra * K + k0 + sel;
      v8bf lo = *(const v8bf*)pa;
      v8bf hi = *(const v8bf*)(pa + 16);
#pragma unroll
      for (int i = 0; i < 8; i++) { a[t][i] = lo[i]; a[t][i + 8] = hi[i]; }
      int rb = rB0 + t * 16; if (rb > N - 1) rb = N - 1;
      const unsigned short* pb = Bt + (size_t)rb * K + k0 + sel;
      v8bf blo = *(const v8bf*)pb;
      v8bf bhi = *(const v8bf*)(pb + 16);
#pragma unroll
      for (int i = 0; i < 8; i++) { b[t][i] = blo[i]; b[t][i + 8] = bhi[i]; }
    }
    if (k0 + 32 < K) {  // gfx1250 global_prefetch_b8 for the next K-slab
      __builtin_prefetch(A + (size_t)rA0 * K + k0 + 32, 0, 0);
      __builtin_prefetch(Bt + (size_t)rB0 * K + k0 + 32, 0, 0);
    }
#pragma unroll
    for (int i = 0; i < 2; i++)
#pragma unroll
      for (int j = 0; j < 2; j++)
        acc[i][j] = __builtin_amdgcn_wmma_f32_16x16x32_bf16(
            false, a[i], false, b[j], (short)0, acc[i][j], false, false);
  }
  const int rowoff = (lane >> 4) << 3;
  const int ncol = lane & 15;
#pragma unroll
  for (int i = 0; i < 2; i++)
#pragma unroll
    for (int j = 0; j < 2; j++)
#pragma unroll
      for (int r = 0; r < 8; r++) {
        int m = m0 + i * 16 + rowoff + r;
        int n = n0 + j * 16 + ncol;
        if (m < M && n < N) {
          float v = acc[i][j][r];
          if (bias && n < Ntrue) v += bias[n];
          if (act == 1) v = geluf(v);
          else if (act == 2) v = v * sigmf(1.702f * v);
          if (resid) v += resid[(size_t)m * N + n];
          if (outF) outF[(size_t)m * N + n] = v;
          if (outB) outB[(size_t)m * N + n] = f2bf(v);
        }
      }
}

// ---------------- MHA (12 heads, d=64, N=513), online softmax ----------------
__global__ __launch_bounds__(128)
void k_mha(const float* __restrict__ qkv, unsigned short* __restrict__ outB) {
  const int bh = blockIdx.x;                 // b*12 + h
  const int b = bh / 12, h = bh % 12;
  const int n = blockIdx.y * 128 + threadIdx.x;
  if (n >= 513) return;
  const float* q = qkv + ((size_t)(b * 513 + n)) * QKN + h * 64;
  float qr[64];
#pragma unroll
  for (int d = 0; d < 64; d++) qr[d] = q[d] * 0.125f;   // d^-0.5
  float mx = -3.4e38f, l = 0.f, o[64];
#pragma unroll
  for (int d = 0; d < 64; d++) o[d] = 0.f;
  for (int m = 0; m < 513; m++) {
    const float* kp = qkv + ((size_t)(b * 513 + m)) * QKN + 768 + h * 64;
    float s = 0.f;
#pragma unroll
    for (int d = 0; d < 64; d++) s += qr[d] * kp[d];
    float nm = fmaxf(mx, s);
    float corr = __expf(mx - nm);
    float e = __expf(s - nm);
    const float* vp = qkv + ((size_t)(b * 513 + m)) * QKN + 1536 + h * 64;
    l = l * corr + e;
#pragma unroll
    for (int d = 0; d < 64; d++) o[d] = o[d] * corr + e * vp[d];
    mx = nm;
  }
  unsigned short* outr = outB + ((size_t)(b * 513 + n)) * CC + h * 64;
  float invl = 1.f / l;
#pragma unroll
  for (int d = 0; d < 64; d++) outr[d] = f2bf(o[d] * invl);
}

// ---------------- tiny view attention (6 heads, dh=2, 128 keys, masked) ----------------
__global__ __launch_bounds__(128)
void k_attn1(const float* __restrict__ qkv1, const unsigned char* __restrict__ mask,
             unsigned short* __restrict__ outB) {
  const int bh = blockIdx.x;            // bc*6 + h
  const int bc = bh / NH1_, h = bh % NH1_;
  const int n = threadIdx.x;
  const int fr = bc * NCL_ + n;
  const float sc = 0.28867513459f;      // 12^-0.5
  const float* qr = qkv1 + (size_t)fr * NPQ + h * 2;
  float q0 = qr[0] * sc, q1 = qr[1] * sc;
  const unsigned char* mrow = mask + ((size_t)bc * NCL_ + n) * NCL_;
  float mx = -3.4e38f;
  for (int m = 0; m < NCL_; m++) {
    const float* kr = qkv1 + (size_t)(bc * NCL_ + m) * NPQ + MID_ + h * 2;
    float s = q0 * kr[0] + q1 * kr[1] + (mrow[m] ? -100000.f : 0.f);
    mx = fmaxf(mx, s);
  }
  float l = 0.f, o0 = 0.f, o1 = 0.f;
  for (int m = 0; m < NCL_; m++) {
    const float* kr = qkv1 + (size_t)(bc * NCL_ + m) * NPQ + MID_ + h * 2;
    float s = q0 * kr[0] + q1 * kr[1] + (mrow[m] ? -100000.f : 0.f);
    float e = __expf(s - mx);
    const float* vr = qkv1 + (size_t)(bc * NCL_ + m) * NPQ + 2 * MID_ + h * 2;
    l += e; o0 += e * vr[0]; o1 += e * vr[1];
  }
  outB[(size_t)fr * KPP + h * 2]     = f2bf(o0 / l);
  outB[(size_t)fr * KPP + h * 2 + 1] = f2bf(o1 / l);
}

// ---------------- elementwise glue ----------------
__global__ void k_combine_inplace(float* __restrict__ x1, const float* __restrict__ h,
                                  const float* __restrict__ adp, size_t n) {
  size_t t = (size_t)blockIdx.x * blockDim.x + threadIdx.x;
  if (t < n) x1[t] = x1[t] + h[t] + 0.1f * adp[t];
}
__global__ void k_copy_cls(const float* __restrict__ x2, float* __restrict__ out) {
  size_t t = (size_t)blockIdx.x * blockDim.x + threadIdx.x;
  if (t >= (size_t)BB * CC) return;
  size_t b = t / CC, c = t % CC;
  out[((size_t)b * 513) * CC + c] = x2[((size_t)b * 513) * CC + c];
}
__global__ void k_gather_feat(const float* __restrict__ x2, float* __restrict__ feat) {
  size_t t = (size_t)blockIdx.x * blockDim.x + threadIdx.x;
  if (t >= (size_t)MF * CC) return;
  size_t fr = t / CC, c = t % CC;
  size_t src = ((fr >> 9) * 513 + 1 + (fr & 511)) * CC + c;
  feat[t] = x2[src];
}

// ---------------- segment max + mean ----------------
__global__ void k_seg_clear(float* sum, unsigned* mx, float* cnt, int ns, int Cn) {
  size_t t = (size_t)blockIdx.x * blockDim.x + threadIdx.x;
  size_t tot = (size_t)ns * Cn;
  if (t < tot) { sum[t] = 0.f; mx[t] = 0u; }
  if (t < (size_t)ns) cnt[t] = 0.f;
}
__global__ void k_seg_scatter(const float* __restrict__ X, const int* __restrict__ idx,
                              float* sum, unsigned* mx, float* cnt, int R, int Cn) {
  size_t t = (size_t)blockIdx.x * blockDim.x + threadIdx.x;
  if (t >= (size_t)R * Cn) return;
  int r = (int)(t / Cn), c = (int)(t % Cn);
  int s = idx[r];
  float v = X[t];
  atomicAdd(&sum[(size_t)s * Cn + c], v);
  atomicMax(&mx[(size_t)s * Cn + c], fenc(v));
  if (c == 0) atomicAdd(&cnt[s], 1.f);
}
__global__ void k_seg_finalize(const float* __restrict__ sum, const unsigned* __restrict__ mx,
                               const float* __restrict__ cnt, float* __restrict__ f,
                               int ns, int Cn) {
  size_t t = (size_t)blockIdx.x * blockDim.x + threadIdx.x;
  if (t >= (size_t)ns * Cn) return;
  int r = (int)(t / Cn);
  float m = fdec(mx[t]);
  if (!__builtin_isfinite(m)) m = 0.f;
  f[t] = m + sum[t] / fmaxf(cnt[r], 1.f);
}

// ---------------- BatchNorm stats (over rows, per channel) ----------------
__global__ __launch_bounds__(256)
void k_bn_stats(const float* __restrict__ f, int rows, int Cn,
                float* __restrict__ mean, float* __restrict__ var) {
  __shared__ float s1[256], s2[256];
  int c = blockIdx.x;
  float s = 0.f, ss = 0.f;
  for (int r = threadIdx.x; r < rows; r += 256) {
    float v = f[(size_t)r * Cn + c]; s += v; ss += v * v;
  }
  s1[threadIdx.x] = s; s2[threadIdx.x] = ss; __syncthreads();
  for (int o = 128; o > 0; o >>= 1) {
    if (threadIdx.x < o) { s1[threadIdx.x] += s1[threadIdx.x + o]; s2[threadIdx.x] += s2[threadIdx.x + o]; }
    __syncthreads();
  }
  if (threadIdx.x == 0) {
    float m = s1[0] / rows;
    mean[c] = m;
    var[c] = s2[0] / rows - m * m;
  }
}

__global__ void k_bn_gelu_gather(const float* __restrict__ f, const int* __restrict__ idx,
                                 const float* __restrict__ mean, const float* __restrict__ var,
                                 const float* __restrict__ g, const float* __restrict__ b,
                                 float* __restrict__ out, int rows, int Cn) {
  size_t t = (size_t)blockIdx.x * blockDim.x + threadIdx.x;
  if (t >= (size_t)rows * Cn) return;
  int r = (int)(t / Cn), c = (int)(t % Cn);
  float v = f[(size_t)idx[r] * Cn + c];
  v = (v - mean[c]) * rsqrtf(var[c] + 1e-5f) * g[c] + b[c];
  out[t] = geluf(v);
}

__global__ __launch_bounds__(256)
void k_rownorm(const float* __restrict__ X, float* __restrict__ nrm, int Cn) {
  __shared__ float s1[256];
  int r = blockIdx.x;
  float s = 0.f;
  for (int c = threadIdx.x; c < Cn; c += 256) { float v = X[(size_t)r * Cn + c]; s += v * v; }
  s1[threadIdx.x] = s; __syncthreads();
  for (int o = 128; o > 0; o >>= 1) {
    if (threadIdx.x < o) s1[threadIdx.x] += s1[threadIdx.x + o];
    __syncthreads();
  }
  if (threadIdx.x == 0) nrm[r] = sqrtf(s1[0]);
}

// dot(view_row, x3d_row) and ||view_row||, view recomputed on the fly
__global__ __launch_bounds__(256)
void k_view_cos(const float* __restrict__ f2, const int* __restrict__ fgi,
                const float* __restrict__ mean, const float* __restrict__ var,
                const float* __restrict__ g, const float* __restrict__ b,
                const float* __restrict__ x3d,
                float* __restrict__ dotv, float* __restrict__ nvv, int Cn) {
  __shared__ float s1[256], s2[256];
  int fr = blockIdx.x;
  int sidx = fgi[fr];
  float d = 0.f, nv = 0.f;
  for (int c = threadIdx.x; c < Cn; c += 256) {
    float v = f2[(size_t)sidx * Cn + c];
    v = (v - mean[c]) * rsqrtf(var[c] + 1e-5f) * g[c] + b[c];
    v = geluf(v);
    d += v * x3d[(size_t)fr * Cn + c];
    nv += v * v;
  }
  s1[threadIdx.x] = d; s2[threadIdx.x] = nv; __syncthreads();
  for (int o = 128; o > 0; o >>= 1) {
    if (threadIdx.x < o) { s1[threadIdx.x] += s1[threadIdx.x + o]; s2[threadIdx.x] += s2[threadIdx.x + o]; }
    __syncthreads();
  }
  if (threadIdx.x == 0) { dotv[fr] = s1[0]; nvv[fr] = sqrtf(s2[0]); }
}

// cosine-weighted view fusion -> output feat rows
__global__ __launch_bounds__(256)
void k_final(const float* __restrict__ feat, const float* __restrict__ f2all,
             const int* __restrict__ fgiall,
             const float* __restrict__ mean2, const float* __restrict__ var2,
             const float* __restrict__ g2, const float* __restrict__ b2,
             const float* __restrict__ dots, const float* __restrict__ nvs,
             const float* __restrict__ nx, float* __restrict__ out, int Cn) {
  int fr = blockIdx.x;
  float w[NV_]; float ssum = 0.f;
  float xn = nx[fr];
#pragma unroll
  for (int i = 0; i < NV_; i++) {
    float cosv = dots[(size_t)i * MF + fr] / fmaxf(nvs[(size_t)i * MF + fr] * xn, 1e-8f);
    w[i] = (cosv + 1.f) * 0.5f;
    ssum += w[i];
  }
#pragma unroll
  for (int i = 0; i < NV_; i++) w[i] /= ssum;
  int sidx[NV_];
#pragma unroll
  for (int i = 0; i < NV_; i++) sidx[i] = fgiall[(size_t)i * MF + fr];
  size_t orow = (((size_t)(fr >> 9)) * 513 + 1 + (fr & 511)) * Cn;
  for (int c = threadIdx.x; c < Cn; c += 256) {
    float acc = 0.f;
#pragma unroll
    for (int i = 0; i < NV_; i++) {
      float v = f2all[(((size_t)i * D2D_) + sidx[i]) * Cn + c];
      v = (v - mean2[i * Cn + c]) * rsqrtf(var2[i * Cn + c] + 1e-5f) * g2[i * Cn + c] + b2[i * Cn + c];
      v = geluf(v);
      acc += w[i] * v;
    }
    out[orow + c] = feat[(size_t)fr * Cn + c] + 0.3f * acc;
  }
}

// ---------------- host ----------------
static inline int cdiv(long a, long b) { return (int)((a + b - 1) / b); }
static inline size_t aup(size_t v) { return (v + 255) & ~(size_t)255; }

extern "C" void kernel_launch(void* const* d_in, const int* in_sizes, int n_in,
                              void* d_out, int out_size, void* d_ws, size_t ws_size,
                              hipStream_t stream) {
  (void)in_sizes; (void)n_in; (void)out_size; (void)ws_size;
  const float* x       = (const float*)d_in[0];
  const int*   cluster = (const int*)d_in[1];
  const int*   fgi     = (const int*)d_in[2];
  const unsigned char* mask = (const unsigned char*)d_in[3];
  const float* g1 = (const float*)d_in[4],  *b1 = (const float*)d_in[5];
  const float* g2 = (const float*)d_in[6],  *b2v = (const float*)d_in[7];
  const float* wqkv = (const float*)d_in[8],  *bqkv = (const float*)d_in[9];
  const float* wproj = (const float*)d_in[10], *bproj = (const float*)d_in[11];
  const float* wfc1 = (const float*)d_in[12], *bfc1 = (const float*)d_in[13];
  const float* wfc2 = (const float*)d_in[14], *bfc2 = (const float*)d_in[15];
  const float* wa1 = (const float*)d_in[16],  *ba1 = (const float*)d_in[17];
  const float* wa2 = (const float*)d_in[18],  *ba2 = (const float*)d_in[19];
  const float* bn3g = (const float*)d_in[20], *bn3b = (const float*)d_in[21];
  const float* g3 = (const float*)d_in[22],   *b3 = (const float*)d_in[23];
  const float* a1wqkv = (const float*)d_in[24], *a1bqkv = (const float*)d_in[25];
  const float* a1wp = (const float*)d_in[26],   *a1bp = (const float*)d_in[27];
  const float* bn2g = (const float*)d_in[28],   *bn2b = (const float*)d_in[29];
  float* out = (float*)d_out;

  // ---- bump-allocate workspace ----
  char* p = (char*)d_ws;
  auto alloc = [&](size_t bytes) { char* r = p; p += aup(bytes); return r; };
  unsigned short* wqkvT  = (unsigned short*)alloc((size_t)QKN * CC * 2);
  unsigned short* wprojT = (unsigned short*)alloc((size_t)CC * CC * 2);
  unsigned short* wfc1T  = (unsigned short*)alloc((size_t)MLPH * CC * 2);
  unsigned short* wfc2T  = (unsigned short*)alloc((size_t)CC * MLPH * 2);
  unsigned short* wa1T   = (unsigned short*)alloc((size_t)ADPN * CC * 2);
  unsigned short* wa2T   = (unsigned short*)alloc((size_t)CC * ADPN * 2);
  unsigned short* vqkvT  = (unsigned short*)alloc((size_t)NV_ * NPQ * CC * 2);
  unsigned short* vwpT   = (unsigned short*)alloc((size_t)NV_ * CC * KPP * 2);
  unsigned short* lnb    = (unsigned short*)alloc((size_t)M1 * CC * 2);
  float*          qkv    = (float*)alloc((size_t)M1 * QKN * 4);
  unsigned short* attnb  = (unsigned short*)alloc((size_t)M1 * CC * 2);
  float*          x1     = (float*)alloc((size_t)M1 * CC * 4);    // becomes x2 in place
  unsigned short* h1b    = (unsigned short*)alloc((size_t)M1 * MLPH * 2);
  float*          hF     = (float*)alloc((size_t)M1 * CC * 4);
  unsigned short* hB     = (unsigned short*)alloc((size_t)M1 * CC * 2);
  unsigned short* tb     = (unsigned short*)alloc((size_t)M1 * ADPN * 2);
  float*          adp    = (float*)alloc((size_t)M1 * CC * 4);    // reused as fx later
  float*          feat   = (float*)alloc((size_t)MF * CC * 4);
  float*          f3     = (float*)alloc((size_t)S3D_ * CC * 4);
  float*          x3d    = (float*)alloc((size_t)MF * CC * 4);
  float*          segsum = (float*)alloc((size_t)D2D_ * CC * 4);
  unsigned*       segmax = (unsigned*)alloc((size_t)D2D_ * CC * 4);
  float*          segcnt = (float*)alloc((size_t)D2D_ * 4);
  float*          mean3  = (float*)alloc((size_t)CC * 4);
  float*          var3   = (float*)alloc((size_t)CC * 4);
  float*          mean2  = (float*)alloc((size_t)NV_ * CC * 4);
  float*          var2   = (float*)alloc((size_t)NV_ * CC * 4);
  float*          qkv1   = (float*)alloc((size_t)MF * NPQ * 4);
  unsigned short* a1outb = (unsigned short*)alloc((size_t)MF * KPP * 2);
  float*          dots   = (float*)alloc((size_t)NV_ * MF * 4);
  float*          nvs    = (float*)alloc((size_t)NV_ * MF * 4);
  float*          nx     = (float*)alloc((size_t)MF * 4);
  float*          f2all  = (float*)qkv;   // alias: qkv (151MB) dead after MHA; f2all needs 134MB
  float*          fx     = adp;           // alias: adp dead after combine

  // ---- weight prep (transpose + bf16 + pad) ----
  k_wt<<<cdiv((long)QKN * CC, 256), 256, 0, stream>>>(wqkv, wqkvT, CC, QKN, CC, QKN);
  k_wt<<<cdiv((long)CC * CC, 256), 256, 0, stream>>>(wproj, wprojT, CC, CC, CC, CC);
  k_wt<<<cdiv((long)MLPH * CC, 256), 256, 0, stream>>>(wfc1, wfc1T, CC, MLPH, CC, MLPH);
  k_wt<<<cdiv((long)CC * MLPH, 256), 256, 0, stream>>>(wfc2, wfc2T, MLPH, CC, MLPH, CC);
  k_wt<<<cdiv((long)ADPN * CC, 256), 256, 0, stream>>>(wa1, wa1T, CC, ADPN, CC, ADPN);
  k_wt<<<cdiv((long)CC * ADPN, 256), 256, 0, stream>>>(wa2, wa2T, ADPN, CC, ADPN, CC);
  for (int i = 0; i < NV_; i++) {
    k_wt<<<cdiv((long)NPQ * CC, 256), 256, 0, stream>>>(
        a1wqkv + (size_t)i * CC * 36, vqkvT + (size_t)i * NPQ * CC, CC, 36, CC, NPQ);
    k_wt<<<cdiv((long)CC * KPP, 256), 256, 0, stream>>>(
        a1wp + (size_t)i * MID_ * CC, vwpT + (size_t)i * CC * KPP, MID_, CC, KPP, CC);
  }
  k_zero_u16<<<cdiv((long)MF * KPP, 256), 256, 0, stream>>>(a1outb, (size_t)MF * KPP);

  dim3 blk128(128), blk256(256);
  // ---- block 1: x = x + MHA(LN(x)) ----
  k_layernorm_bf16<<<M1, blk256, 0, stream>>>(x, g1, b1, lnb, CC);
  k_gemm_wmma<<<dim3(cdiv(QKN, 64), cdiv(M1, 64)), blk128, 0, stream>>>(
      lnb, wqkvT, bqkv, nullptr, qkv, nullptr, M1, QKN, CC, QKN, 0);
  k_mha<<<dim3(BB * 12, 5), blk128, 0, stream>>>(qkv, attnb);
  k_gemm_wmma<<<dim3(cdiv(CC, 64), cdiv(M1, 64)), blk128, 0, stream>>>(
      attnb, wprojT, bproj, x, x1, nullptr, M1, CC, CC, CC, 0);

  // ---- block 2: MLP + adapter ----
  k_layernorm_bf16<<<M1, blk256, 0, stream>>>(x1, g2, b2v, lnb, CC);
  k_gemm_wmma<<<dim3(cdiv(MLPH, 64), cdiv(M1, 64)), blk128, 0, stream>>>(
      lnb, wfc1T, bfc1, nullptr, nullptr, h1b, M1, MLPH, CC, MLPH, 1);
  k_gemm_wmma<<<dim3(cdiv(CC, 64), cdiv(M1, 64)), blk128, 0, stream>>>(
      h1b, wfc2T, bfc2, nullptr, hF, hB, M1, CC, MLPH, CC, 0);
  k_gemm_wmma<<<dim3(cdiv(ADPN, 64), cdiv(M1, 64)), blk128, 0, stream>>>(
      hB, wa1T, ba1, nullptr, nullptr, tb, M1, ADPN, CC, ADPN, 2);
  k_gemm_wmma<<<dim3(cdiv(CC, 64), cdiv(M1, 64)), blk128, 0, stream>>>(
      tb, wa2T, ba2, nullptr, adp, nullptr, M1, CC, ADPN, CC, 0);
  k_combine_inplace<<<cdiv((long)M1 * CC, 256), 256, 0, stream>>>(x1, hF, adp, (size_t)M1 * CC);

  // x1 now holds x2; emit cls rows and gather feat
  k_copy_cls<<<cdiv((long)BB * CC, 256), 256, 0, stream>>>(x1, out);
  k_gather_feat<<<cdiv((long)MF * CC, 256), 256, 0, stream>>>(x1, feat);

  // ---- 3D branch ----
  k_seg_clear<<<cdiv((long)S3D_ * CC, 256), 256, 0, stream>>>(segsum, segmax, segcnt, S3D_, CC);
  k_seg_scatter<<<cdiv((long)MF * CC, 256), 256, 0, stream>>>(feat, cluster, segsum, segmax, segcnt, MF, CC);
  k_seg_finalize<<<cdiv((long)S3D_ * CC, 256), 256, 0, stream>>>(segsum, segmax, segcnt, f3, S3D_, CC);
  k_bn_stats<<<CC, blk256, 0, stream>>>(f3, S3D_, CC, mean3, var3);
  k_bn_gelu_gather<<<cdiv((long)MF * CC, 256), 256, 0, stream>>>(
      f3, cluster, mean3, var3, bn3g, bn3b, x3d, MF, CC);
  k_rownorm<<<MF, blk256, 0, stream>>>(x3d, nx, CC);

  // ---- view branches ----
  for (int i = 0; i < NV_; i++) {
    const int* fgi_i = fgi + (size_t)i * MF;
    k_layernorm_bf16<<<MF, blk256, 0, stream>>>(feat, g3 + (size_t)i * CC, b3 + (size_t)i * CC, lnb, CC);
    k_gemm_wmma<<<dim3(cdiv(NPQ, 64), cdiv(MF, 64)), blk128, 0, stream>>>(
        lnb, vqkvT + (size_t)i * NPQ * CC, a1bqkv + (size_t)i * 36, nullptr,
        qkv1, nullptr, MF, NPQ, CC, 36, 0);
    k_attn1<<<MF / NCL_ * NH1_, blk128, 0, stream>>>(
        qkv1, mask + (size_t)i * (MF / NCL_) * NCL_ * NCL_, a1outb);
    k_gemm_wmma<<<dim3(cdiv(CC, 64), cdiv(MF, 64)), blk128, 0, stream>>>(
        a1outb, vwpT + (size_t)i * CC * KPP, a1bp + (size_t)i * CC, feat,
        fx, nullptr, MF, CC, KPP, CC, 0);
    float* f2i = f2all + (size_t)i * D2D_ * CC;
    k_seg_clear<<<cdiv((long)D2D_ * CC, 256), 256, 0, stream>>>(segsum, segmax, segcnt, D2D_, CC);
    k_seg_scatter<<<cdiv((long)MF * CC, 256), 256, 0, stream>>>(fx, fgi_i, segsum, segmax, segcnt, MF, CC);
    k_seg_finalize<<<cdiv((long)D2D_ * CC, 256), 256, 0, stream>>>(segsum, segmax, segcnt, f2i, D2D_, CC);
    k_bn_stats<<<CC, blk256, 0, stream>>>(f2i, D2D_, CC, mean2 + (size_t)i * CC, var2 + (size_t)i * CC);
    k_view_cos<<<MF, blk256, 0, stream>>>(
        f2i, fgi_i, mean2 + (size_t)i * CC, var2 + (size_t)i * CC,
        bn2g + (size_t)i * CC, bn2b + (size_t)i * CC, x3d,
        dots + (size_t)i * MF, nvs + (size_t)i * MF, CC);
  }

  // ---- cosine-weighted fusion -> output feat rows ----
  k_final<<<MF, blk256, 0, stream>>>(feat, f2all, fgi, mean2, var2, bn2g, bn2b,
                                     dots, nvs, nx, out, CC);
}
